// GPTAttention_79963701117094
// MI455X (gfx1250) — compile-verified, matrix operand order checked by reading, not run on hardware
//
#include <hip/hip_runtime.h>
#include <hip/hip_bf16.h>

// ---------------------------------------------------------------------------
// GPT causal attention forward for MI455X (gfx1250, wave32, WMMA).
// All GEMM-shaped math (QKV proj, Q·K^T, P·V, out proj) goes through
// v_wmma_f32_16x16x32_f16 (f16 inputs, f32 accumulate).
// Global->LDS GEMM staging uses CDNA5 async copies
// (global_load_async_to_lds_b128, ASYNCcnt) + global_prefetch hints.
// ---------------------------------------------------------------------------

typedef __attribute__((ext_vector_type(16))) _Float16 v16h;
typedef __attribute__((ext_vector_type(8)))  float    v8f;

#define DM     768
#define NHEADS 12
#define HD     64
#define SEQ    2048
#define BATCH  4
#define ROWS   (BATCH * SEQ)      // 8192

// GEMM tiling
#define BM  128
#define BN  64
#define BK  32
#define LDA (BK + 8)              // LDS pitch in halfs (80B, 16B-aligned rows)

union FragU {
  v16h v;
  _Float16 h[16];
  uint4 q[2];
};

// Load a 16x32 f16 WMMA A-fragment (or a B-fragment from an N-major matrix)
// from a row-major half matrix. Per ISA 7.12.2 layout:
//   lane L: row = L&15; chunk offset ko = (L>=16)?8:0;
//   elements 0..7  -> k = k0+ko .. k0+ko+7
//   elements 8..15 -> k = k0+ko+16 .. k0+ko+23
__device__ __forceinline__ v16h load_frag_rm(const _Float16* __restrict__ base,
                                             int row0, int ld, int k0, int lane) {
  const int m  = lane & 15;
  const int ko = (lane >> 4) << 3;
  const _Float16* p = base + (size_t)(row0 + m) * ld + k0 + ko;
  FragU f;
  f.q[0] = *(const uint4*)(p);
  f.q[1] = *(const uint4*)(p + 16);
  return f.v;
}

__device__ __forceinline__ v8f wmma_f16(v16h a, v16h b, v8f c) {
  // (neg_a, A, neg_b, B, c_mod, C, reuse_a, reuse_b)
  return __builtin_amdgcn_wmma_f32_16x16x32_f16(false, a, false, b, (short)0, c,
                                                false, false);
}

__device__ __forceinline__ v8f v8f_zero() {
  v8f z = {0.f, 0.f, 0.f, 0.f, 0.f, 0.f, 0.f, 0.f};
  return z;
}

// CDNA5 async global->LDS copy (16 bytes per lane). The LDS operand is the
// wave-relative LDS address = low 32 bits of the generic pointer (flat
// addresses in the LDS aperture truncate to the LDS offset, ISA 10.2).
// Tracked by ASYNCcnt; pair with s_wait_asynccnt before the barrier.
__device__ __forceinline__ void async_ld_b128(void* lds_ptr, const void* gptr) {
  unsigned loff = (unsigned)(size_t)lds_ptr;
  asm volatile("global_load_async_to_lds_b128 %0, %1, off"
               :: "v"(loff), "v"(gptr)
               : "memory");
}
__device__ __forceinline__ void wait_async0() {
  asm volatile("s_wait_asynccnt 0x0" ::: "memory");
}

// ---------------------------------------------------------------------------
// Kernel 1: f32 -> f16 elementwise convert (for x)
// ---------------------------------------------------------------------------
__global__ void cvt_f2h_kernel(const float* __restrict__ src,
                               _Float16* __restrict__ dst, int n) {
  int i = blockIdx.x * 256 + threadIdx.x;
  if (i < n) dst[i] = (_Float16)src[i];
}

// ---------------------------------------------------------------------------
// Kernel 2: transpose + convert weights: Wt[o][i] = (f16) W[i][o]
// blockIdx.z selects which of the 4 weights.
// ---------------------------------------------------------------------------
__global__ void xpose_w_kernel(const float* __restrict__ w0, const float* __restrict__ w1,
                               const float* __restrict__ w2, const float* __restrict__ w3,
                               _Float16* __restrict__ t0, _Float16* __restrict__ t1,
                               _Float16* __restrict__ t2, _Float16* __restrict__ t3) {
  const int z = blockIdx.z;
  const float* src = (z == 0) ? w0 : (z == 1) ? w1 : (z == 2) ? w2 : w3;
  _Float16*    dst = (z == 0) ? t0 : (z == 1) ? t1 : (z == 2) ? t2 : t3;
  int i = blockIdx.x * 256 + threadIdx.x;
  if (i < DM * DM) {
    int o = i / DM, k = i - o * DM;
    dst[(size_t)o * DM + k] = (_Float16)src[(size_t)k * DM + o];
  }
}

// ---------------------------------------------------------------------------
// Kernel 3: QKV projection GEMM.
//   A = xh [8192 x 768] (row-major f16), B = Wt [768(out) x 768(in)] f16.
//   blockIdx.z: 0=Q, 1=K, 2=V.
//   Q,K stored [B,H,S,64]; V stored transposed [B,H,64,S].
// 256 threads = 8 waves; block tile 128x64; wave tile 32x32 (2x2 WMMA tiles).
// ---------------------------------------------------------------------------
__global__ __launch_bounds__(256) void qkv_gemm_kernel(
    const _Float16* __restrict__ xh,
    const _Float16* __restrict__ WtQ, const _Float16* __restrict__ WtK,
    const _Float16* __restrict__ WtV,
    const float* __restrict__ bq, const float* __restrict__ bk,
    const float* __restrict__ bv,
    _Float16* __restrict__ Qh, _Float16* __restrict__ Kh,
    _Float16* __restrict__ Vth) {
  const int t = blockIdx.z;
  const _Float16* Wt  = (t == 0) ? WtQ : (t == 1) ? WtK : WtV;
  const float*   bias = (t == 0) ? bq  : (t == 1) ? bk  : bv;

  __shared__ alignas(16) _Float16 lds_a[BM][LDA];
  __shared__ alignas(16) _Float16 lds_b[BN][LDA];

  const int tid  = threadIdx.x;
  const int lane = tid & 31;
  const int wave = tid >> 5;
  const int wm   = wave >> 1;  // 0..3
  const int wn   = wave & 1;   // 0..1
  const int row0 = blockIdx.y * BM;
  const int col0 = blockIdx.x * BN;

  const int ar = tid >> 1;            // A tile row this thread stages
  const int ak = (tid & 1) * 16;      // A tile k-offset (halfs)
  const int bn = tid >> 2;            // B tile row (out-feature)
  const int bk2 = (tid & 3) * 8;      // B tile k-offset (halfs)

  v8f c00 = v8f_zero(), c01 = v8f_zero(), c10 = v8f_zero(), c11 = v8f_zero();

  for (int k0 = 0; k0 < DM; k0 += BK) {
    // Stage A tile (128x32 halfs) and B tile (64x32 halfs) via async copies.
    {
      const _Float16* g = xh + (size_t)(row0 + ar) * DM + k0 + ak;
      async_ld_b128(&lds_a[ar][ak],     g);
      async_ld_b128(&lds_a[ar][ak + 8], g + 8);
      async_ld_b128(&lds_b[bn][bk2],
                    Wt + (size_t)(col0 + bn) * DM + k0 + bk2);
    }
    if (k0 + BK < DM) {  // prefetch next K tile into cache
      __builtin_prefetch(xh + (size_t)(row0 + ar) * DM + k0 + BK, 0, 0);
      __builtin_prefetch(Wt + (size_t)(col0 + bn) * DM + k0 + BK, 0, 0);
    }
    wait_async0();
    __syncthreads();

    v16h a0 = load_frag_rm(&lds_a[0][0], wm * 32 + 0,  LDA, 0, lane);
    v16h a1 = load_frag_rm(&lds_a[0][0], wm * 32 + 16, LDA, 0, lane);
    v16h b0 = load_frag_rm(&lds_b[0][0], wn * 32 + 0,  LDA, 0, lane);
    v16h b1 = load_frag_rm(&lds_b[0][0], wn * 32 + 16, LDA, 0, lane);

    c00 = wmma_f16(a0, b0, c00);
    c01 = wmma_f16(a0, b1, c01);
    c10 = wmma_f16(a1, b0, c10);
    c11 = wmma_f16(a1, b1, c11);
    __syncthreads();
  }

  // Epilogue: bias + scatter to head-major layouts.
  const int hi = (lane >> 4) << 3;  // row offset 0 or 8 within 16x16 tile
  const int cn = lane & 15;
  v8f cf[2][2] = {{c00, c01}, {c10, c11}};
  for (int mt = 0; mt < 2; ++mt) {
    for (int nt = 0; nt < 2; ++nt) {
      for (int e = 0; e < 8; ++e) {
        int grow = row0 + wm * 32 + mt * 16 + hi + e;   // 0..8191
        int gcol = col0 + wn * 32 + nt * 16 + cn;       // 0..767
        float val = cf[mt][nt][e] + bias[gcol];
        int b = grow >> 11, s = grow & (SEQ - 1);
        int h = gcol >> 6,  d = gcol & (HD - 1);
        size_t bh = (size_t)(b * NHEADS + h);
        if (t == 2) {
          Vth[(bh * HD + d) * SEQ + s] = (_Float16)val;  // [B,H,64,S]
        } else {
          _Float16* dst = (t == 0) ? Qh : Kh;
          dst[(bh * SEQ + s) * HD + d] = (_Float16)val;  // [B,H,S,64]
        }
      }
    }
  }
}

// ---------------------------------------------------------------------------
// Kernel 4: causal flash attention.
// grid (S/64, B*H), 128 threads = 4 waves; each wave owns 16 query rows.
// ---------------------------------------------------------------------------
__global__ __launch_bounds__(128) void attn_kernel(
    const _Float16* __restrict__ Qh, const _Float16* __restrict__ Kh,
    const _Float16* __restrict__ Vth, _Float16* __restrict__ ctxh) {
  const int bh   = blockIdx.y;              // 0..47
  const int b    = bh / NHEADS;
  const int h    = bh - b * NHEADS;
  const int q0   = blockIdx.x * 64;
  const int lane = threadIdx.x & 31;
  const int wave = threadIdx.x >> 5;
  const int hi   = (lane >> 4) << 3;
  const int cn   = lane & 15;
  const int qrow0 = q0 + wave * 16;

  const _Float16* Qb = Qh  + (size_t)bh * SEQ * HD;
  const _Float16* Kb = Kh  + (size_t)bh * SEQ * HD;
  const _Float16* Vb = Vth + (size_t)bh * HD * SEQ;

  __shared__ alignas(16) _Float16 lds_p[4][16][72];  // per-wave P tile (C->A relayout)

  // Q A-frags held in registers for the whole block of queries.
  v16h qf0 = load_frag_rm(Qb, qrow0, HD, 0,  lane);
  v16h qf1 = load_frag_rm(Qb, qrow0, HD, 32, lane);

  float mrow[8], lrow[8];
  v8f acc[4];
#pragma unroll
  for (int e = 0; e < 8; ++e) { mrow[e] = -1e30f; lrow[e] = 0.f; }
#pragma unroll
  for (int nd = 0; nd < 4; ++nd) acc[nd] = v8f_zero();

  const float scale = 0.125f;  // 1/sqrt(64)

  for (int kv0 = 0; kv0 <= q0; kv0 += 64) {
    if (kv0 + 64 <= q0) {  // prefetch next KV block (K rows + V rows)
      __builtin_prefetch(Kb + (size_t)(kv0 + 64 + lane) * HD, 0, 0);
      __builtin_prefetch(Vb + (size_t)(lane << 1) * SEQ + kv0 + 64, 0, 0);
    }
    // ---- scores S = Q . K^T  (16 x 64 per wave) ----
    float pm[4][8];
    const bool diag = (kv0 == q0);
#pragma unroll
    for (int nt = 0; nt < 4; ++nt) {
      v16h kf0 = load_frag_rm(Kb, kv0 + nt * 16, HD, 0,  lane);
      v16h kf1 = load_frag_rm(Kb, kv0 + nt * 16, HD, 32, lane);
      v8f sc = v8f_zero();
      sc = wmma_f16(qf0, kf0, sc);
      sc = wmma_f16(qf1, kf1, sc);
#pragma unroll
      for (int e = 0; e < 8; ++e) {
        float v = sc[e] * scale;
        if (diag) {
          int row = qrow0 + hi + e;
          int col = kv0 + nt * 16 + cn;
          if (col > row) v = -1e30f;
        }
        pm[nt][e] = v;
      }
    }

    // ---- online softmax (row reductions over 16-lane groups) ----
    float mnew[8], alpha[8];
#pragma unroll
    for (int e = 0; e < 8; ++e) {
      float mt = fmaxf(fmaxf(pm[0][e], pm[1][e]), fmaxf(pm[2][e], pm[3][e]));
#pragma unroll
      for (int off = 1; off < 16; off <<= 1)
        mt = fmaxf(mt, __shfl_xor(mt, off, 32));
      mnew[e]  = fmaxf(mrow[e], mt);
      alpha[e] = __expf(mrow[e] - mnew[e]);
      mrow[e]  = mnew[e];
    }

    __syncthreads();  // previous iteration's P-frag reads done (uniform trip count)
#pragma unroll
    for (int nt = 0; nt < 4; ++nt) {
#pragma unroll
      for (int e = 0; e < 8; ++e) {
        float p = __expf(pm[nt][e] - mnew[e]);
        pm[nt][e] = p;
        lds_p[wave][hi + e][nt * 16 + cn] = (_Float16)p;
      }
    }
#pragma unroll
    for (int e = 0; e < 8; ++e) {
      float s = (pm[0][e] + pm[1][e]) + (pm[2][e] + pm[3][e]);
#pragma unroll
      for (int off = 1; off < 16; off <<= 1)
        s += __shfl_xor(s, off, 32);
      lrow[e] = lrow[e] * alpha[e] + s;
    }
#pragma unroll
    for (int nd = 0; nd < 4; ++nd)
#pragma unroll
      for (int e = 0; e < 8; ++e) acc[nd][e] *= alpha[e];
    __syncthreads();

    // ---- ctx += P . V  (P from LDS in A-layout; V transposed -> contiguous) ----
    v16h pf0 = load_frag_rm(&lds_p[wave][0][0], 0, 72, 0,  lane);
    v16h pf1 = load_frag_rm(&lds_p[wave][0][0], 0, 72, 32, lane);
#pragma unroll
    for (int nd = 0; nd < 4; ++nd) {
      v16h vf0 = load_frag_rm(Vb, nd * 16, SEQ, kv0,      lane);
      v16h vf1 = load_frag_rm(Vb, nd * 16, SEQ, kv0 + 32, lane);
      acc[nd] = wmma_f16(pf0, vf0, acc[nd]);
      acc[nd] = wmma_f16(pf1, vf1, acc[nd]);
    }
  }

  // ---- finalize: ctx = acc / l, store f16 [B,S,768] ----
#pragma unroll
  for (int nd = 0; nd < 4; ++nd) {
#pragma unroll
    for (int e = 0; e < 8; ++e) {
      float v = acc[nd][e] / lrow[e];
      int srow = qrow0 + hi + e;
      int col  = h * HD + nd * 16 + cn;
      ctxh[((size_t)(b * SEQ + srow)) * DM + col] = (_Float16)v;
    }
  }
}

// ---------------------------------------------------------------------------
// Kernel 5: output projection GEMM -> f32 d_out.
// ---------------------------------------------------------------------------
__global__ __launch_bounds__(256) void oproj_gemm_kernel(
    const _Float16* __restrict__ ctxh, const _Float16* __restrict__ WtO,
    const float* __restrict__ bo, float* __restrict__ out) {
  __shared__ alignas(16) _Float16 lds_a[BM][LDA];
  __shared__ alignas(16) _Float16 lds_b[BN][LDA];

  const int tid  = threadIdx.x;
  const int lane = tid & 31;
  const int wave = tid >> 5;
  const int wm   = wave >> 1;
  const int wn   = wave & 1;
  const int row0 = blockIdx.y * BM;
  const int col0 = blockIdx.x * BN;

  const int ar  = tid >> 1;
  const int ak  = (tid & 1) * 16;
  const int bn  = tid >> 2;
  const int bk2 = (tid & 3) * 8;

  v8f c00 = v8f_zero(), c01 = v8f_zero(), c10 = v8f_zero(), c11 = v8f_zero();

  for (int k0 = 0; k0 < DM; k0 += BK) {
    {
      const _Float16* g = ctxh + (size_t)(row0 + ar) * DM + k0 + ak;
      async_ld_b128(&lds_a[ar][ak],     g);
      async_ld_b128(&lds_a[ar][ak + 8], g + 8);
      async_ld_b128(&lds_b[bn][bk2],
                    WtO + (size_t)(col0 + bn) * DM + k0 + bk2);
    }
    if (k0 + BK < DM) {
      __builtin_prefetch(ctxh + (size_t)(row0 + ar) * DM + k0 + BK, 0, 0);
      __builtin_prefetch(WtO + (size_t)(col0 + bn) * DM + k0 + BK, 0, 0);
    }
    wait_async0();
    __syncthreads();

    v16h a0 = load_frag_rm(&lds_a[0][0], wm * 32 + 0,  LDA, 0, lane);
    v16h a1 = load_frag_rm(&lds_a[0][0], wm * 32 + 16, LDA, 0, lane);
    v16h b0 = load_frag_rm(&lds_b[0][0], wn * 32 + 0,  LDA, 0, lane);
    v16h b1 = load_frag_rm(&lds_b[0][0], wn * 32 + 16, LDA, 0, lane);

    c00 = wmma_f16(a0, b0, c00);
    c01 = wmma_f16(a0, b1, c01);
    c10 = wmma_f16(a1, b0, c10);
    c11 = wmma_f16(a1, b1, c11);
    __syncthreads();
  }

  const int hi = (lane >> 4) << 3;
  const int cn = lane & 15;
  v8f cf[2][2] = {{c00, c01}, {c10, c11}};
  for (int mt = 0; mt < 2; ++mt) {
    for (int nt = 0; nt < 2; ++nt) {
      for (int e = 0; e < 8; ++e) {
        int grow = row0 + wm * 32 + mt * 16 + hi + e;
        int gcol = col0 + wn * 32 + nt * 16 + cn;
        out[(size_t)grow * DM + gcol] = cf[mt][nt][e] + bo[gcol];
      }
    }
  }
}

// ---------------------------------------------------------------------------
// Host-side launch.
// Inputs (f32): x, Wq, bq, Wk, bk, Wv, bv, Wo, bo.
// ---------------------------------------------------------------------------
extern "C" void kernel_launch(void* const* d_in, const int* in_sizes, int n_in,
                              void* d_out, int out_size, void* d_ws,
                              size_t ws_size, hipStream_t stream) {
  const float* x  = (const float*)d_in[0];
  const float* Wq = (const float*)d_in[1];
  const float* bq = (const float*)d_in[2];
  const float* Wk = (const float*)d_in[3];
  const float* bk = (const float*)d_in[4];
  const float* Wv = (const float*)d_in[5];
  const float* bv = (const float*)d_in[6];
  const float* Wo = (const float*)d_in[7];
  const float* bo = (const float*)d_in[8];
  float* out = (float*)d_out;

  char* ws = (char*)d_ws;
  size_t off = 0;
  auto alloc = [&](size_t bytes) -> void* {
    void* p = ws + off;
    off = (off + bytes + 255) & ~(size_t)255;
    return p;
  };

  const size_t nX  = (size_t)ROWS * DM;   // 8192*768
  const size_t nW  = (size_t)DM * DM;     // 768*768
  const size_t nQK = (size_t)BATCH * NHEADS * SEQ * HD;

  _Float16* xh   = (_Float16*)alloc(nX * 2);
  _Float16* WtQ  = (_Float16*)alloc(nW * 2);
  _Float16* WtK  = (_Float16*)alloc(nW * 2);
  _Float16* WtV  = (_Float16*)alloc(nW * 2);
  _Float16* WtO  = (_Float16*)alloc(nW * 2);
  _Float16* Qh   = (_Float16*)alloc(nQK * 2);
  _Float16* Kh   = (_Float16*)alloc(nQK * 2);
  _Float16* Vth  = (_Float16*)alloc(nQK * 2);
  _Float16* ctxh = (_Float16*)alloc(nX * 2);
  (void)ws_size; (void)in_sizes; (void)n_in; (void)out_size;

  // 1) convert x
  cvt_f2h_kernel<<<(int)((nX + 255) / 256), 256, 0, stream>>>(x, xh, (int)nX);
  // 2) transpose+convert weights
  xpose_w_kernel<<<dim3((DM * DM + 255) / 256, 1, 4), 256, 0, stream>>>(
      Wq, Wk, Wv, Wo, WtQ, WtK, WtV, WtO);
  // 3) QKV projections
  qkv_gemm_kernel<<<dim3(DM / BN, ROWS / BM, 3), 256, 0, stream>>>(
      xh, WtQ, WtK, WtV, bq, bk, bv, Qh, Kh, Vth);
  // 4) flash attention
  attn_kernel<<<dim3(SEQ / 64, BATCH * NHEADS), 128, 0, stream>>>(Qh, Kh, Vth,
                                                                  ctxh);
  // 5) output projection
  oproj_gemm_kernel<<<dim3(DM / BN, ROWS / BM), 256, 0, stream>>>(ctxh, WtO, bo,
                                                                  out);
}
